// PhysicalMoE_35966056137152
// MI455X (gfx1250) — matrix operand order repaired and checked by reading, not compiled, not run on hardware
//
#include <hip/hip_runtime.h>
#include <hip/hip_bf16.h>
#include <math.h>

#define B_TOK   16384
#define IN_DIM  1024
#define HID_DIM 4096
#define OUT_DIM 1024
#define NEXP    8
#define SIG     32
#define RH      16      // router hidden = 2*E

typedef __attribute__((ext_vector_type(16))) __bf16 v16bf;
typedef __attribute__((ext_vector_type(8)))  float  v8f;

__device__ __forceinline__ float gelu_exact(float x) {
  return 0.5f * x * (1.0f + erff(x * 0.70710678118654752f));
}

// ---------------------------------------------------------------------------
// Transpose + f32->bf16 convert:  in [E][R][C] f32  ->  out [E][C][R] bf16
// ---------------------------------------------------------------------------
template <int R, int C>
__global__ void transpose_cvt_kernel(const float* __restrict__ in,
                                     __bf16* __restrict__ out) {
  __shared__ float tile[32][33];
  const int e  = blockIdx.z;
  const int c0 = blockIdx.x * 32, r0 = blockIdx.y * 32;
  const float* ine  = in  + (size_t)e * R * C;
  __bf16*      oute = out + (size_t)e * R * C;
  for (int rr = threadIdx.y; rr < 32; rr += 8)
    tile[rr][threadIdx.x] = ine[(size_t)(r0 + rr) * C + c0 + threadIdx.x];
  __syncthreads();
  for (int cc = threadIdx.y; cc < 32; cc += 8)
    oute[(size_t)(c0 + cc) * R + r0 + threadIdx.x] = (__bf16)tile[threadIdx.x][cc];
}

// ---------------------------------------------------------------------------
// Router: h = gelu([x;sig;tc;rs] @ w1 + b1); logits = h @ w2 + b2; softmax;
// top-2 -> compacted per-expert token lists.
// Block = 128 threads = 8 tokens x 16 hidden units.
// ---------------------------------------------------------------------------
__global__ __launch_bounds__(128)
void router_kernel(const float* __restrict__ x, const float* __restrict__ sig,
                   const float* __restrict__ tc, const float* __restrict__ rs,
                   const float* __restrict__ w1, const float* __restrict__ b1,
                   const float* __restrict__ w2, const float* __restrict__ b2,
                   int* __restrict__ counts, int* __restrict__ tokIds,
                   float* __restrict__ tokW) {
  __shared__ float hh[8][RH];
  const int lt = threadIdx.x >> 4;   // local token 0..7
  const int u  = threadIdx.x & 15;   // hidden unit
  const int t  = blockIdx.x * 8 + lt;
  const float* xr = x + (size_t)t * IN_DIM;
  float s = b1[u];
  for (int i = 0; i < IN_DIM; ++i) s += xr[i] * w1[i * RH + u];
  const float* sr = sig + (size_t)t * SIG;
  for (int i = 0; i < SIG; ++i) s += sr[i] * w1[(IN_DIM + i) * RH + u];
  s += tc[t]        * w1[(IN_DIM + SIG + 0) * RH + u];
  s += rs[2 * t]    * w1[(IN_DIM + SIG + 1) * RH + u];
  s += rs[2 * t + 1]* w1[(IN_DIM + SIG + 2) * RH + u];
  hh[lt][u] = gelu_exact(s);
  __syncthreads();
  if (threadIdx.x < 8) {
    const int tt = blockIdx.x * 8 + threadIdx.x;
    float logit[NEXP];
    float m = -3.4e38f;
    for (int e = 0; e < NEXP; ++e) {
      float l = b2[e];
      for (int q = 0; q < RH; ++q) l += hh[threadIdx.x][q] * w2[q * NEXP + e];
      logit[e] = l; m = fmaxf(m, l);
    }
    float sum = 0.f;
    for (int e = 0; e < NEXP; ++e) { logit[e] = expf(logit[e] - m); sum += logit[e]; }
    const float inv = 1.0f / sum;
    float w0v = -1.f, w1v = -1.f; int i0 = 0, i1 = 0;
    for (int e = 0; e < NEXP; ++e) {        // strict '>' => ties pick lower idx
      float wv = logit[e] * inv;
      if (wv > w0v)      { w1v = w0v; i1 = i0; w0v = wv; i0 = e; }
      else if (wv > w1v) { w1v = wv; i1 = e; }
    }
    int s0 = atomicAdd(&counts[i0], 1);
    tokIds[i0 * B_TOK + s0] = tt; tokW[i0 * B_TOK + s0] = w0v;
    int s1 = atomicAdd(&counts[i1], 1);
    tokIds[i1 * B_TOK + s1] = tt; tokW[i1 * B_TOK + s1] = w1v;
  }
}

// ---------------------------------------------------------------------------
// Fused expert kernel. One block = 32 routed tokens of one expert.
// LDS: xs = 32x1024 bf16 (64 KB), hs = 32x4096 bf16 (256 KB), both stored
// row-major with a 16-byte-chunk XOR swizzle: chunk' = chunk ^ (row & 15).
// ---------------------------------------------------------------------------
__device__ __forceinline__ v16bf lds_a_frag(const char* base, int rowStrideB,
                                            int msub, int k0, int lane) {
  // ISA 16-bit A 16x32: lanes 0-15: M=lane, K = k0+[0..7] and k0+[16..23];
  //                     lanes16-31: M=lane-16, K = k0+[8..15] and k0+[24..31].
  const int m  = (msub << 4) + (lane & 15);
  const int c1 = (k0 >> 3) + (lane >> 4);
  const int sw = m & 15;
  const char* rb = base + m * rowStrideB;
  union { uint4 q[2]; v16bf v; } u;
  u.q[0] = *(const uint4*)(rb + (((c1    ) ^ sw) << 4));
  u.q[1] = *(const uint4*)(rb + (((c1 + 2) ^ sw) << 4));
  return u.v;
}

__global__ __launch_bounds__(256)
void expert_kernel(const float* __restrict__ x,
                   const __bf16* __restrict__ w1T,   // [E][HID][IN] bf16
                   const __bf16* __restrict__ w2T,   // [E][OUT][HID] bf16
                   const float* __restrict__ eb1,    // [E][HID]
                   const float* __restrict__ eb2,    // [E][OUT]
                   const int* __restrict__ counts,
                   const int* __restrict__ tokIds,
                   const float* __restrict__ tokW,
                   float* __restrict__ out) {
  extern __shared__ char smem[];
  char* xs = smem;            // 65536 B
  char* hs = smem + 65536;    // 262144 B
  const int e   = blockIdx.y;
  const int cnt = counts[e];
  const int m0  = blockIdx.x * 32;
  if (m0 >= cnt) return;
  const int rowsValid = min(32, cnt - m0);
  const int tid  = threadIdx.x;
  const int wv   = tid >> 5;
  const int lane = tid & 31;
  const int*   tIds = tokIds + e * B_TOK + m0;
  const float* tW   = tokW   + e * B_TOK + m0;
  const __bf16* w1e = w1T + (size_t)e * HID_DIM * IN_DIM;
  const __bf16* w2e = w2T + (size_t)e * OUT_DIM * HID_DIM;
  const float*  b1e = eb1 + (size_t)e * HID_DIM;
  const float*  b2e = eb2 + (size_t)e * OUT_DIM;

  // ---- Stage 0: gather routed x rows -> LDS bf16 (zero-pad tail rows)
  for (int i = tid; i < 32 * 128; i += 256) {
    const int r = i >> 7, c = i & 127;      // 128 chunks of 8 bf16 per row
    union { uint4 q; __bf16 h[8]; } u;
    if (r < rowsValid) {
      const float* src = x + (size_t)tIds[r] * IN_DIM + (c << 3);
      float4 f0 = *(const float4*)src;
      float4 f1 = *(const float4*)(src + 4);
      u.h[0]=(__bf16)f0.x; u.h[1]=(__bf16)f0.y; u.h[2]=(__bf16)f0.z; u.h[3]=(__bf16)f0.w;
      u.h[4]=(__bf16)f1.x; u.h[5]=(__bf16)f1.y; u.h[6]=(__bf16)f1.z; u.h[7]=(__bf16)f1.w;
    } else {
      u.q = make_uint4(0u, 0u, 0u, 0u);
    }
    *(uint4*)(xs + r * 2048 + ((c ^ (r & 15)) << 4)) = u.q;
  }
  __syncthreads();

  // ---- Stage 1: hs = gelu(xs @ W1 + b1).  8 waves x 32 n-tiles = 256 tiles.
  for (int ntg = 0; ntg < 8; ++ntg) {
    const int ntBase = wv * 32 + ntg * 4;
    v8f acc[4][2] = {};
    for (int kk = 0; kk < 32; ++kk) {
      const int k0 = kk << 5;
      const v16bf a0 = lds_a_frag(xs, 2048, 0, k0, lane);
      const v16bf a1 = lds_a_frag(xs, 2048, 1, k0, lane);
#pragma unroll
      for (int j = 0; j < 4; ++j) {
        const int n = (ntBase + j) << 4;
        const __bf16* bp = w1e + (size_t)(n + (lane & 15)) * IN_DIM + k0 + ((lane >> 4) << 3);
        union { uint4 q[2]; v16bf v; } ub;
        ub.q[0] = *(const uint4*)bp; ub.q[1] = *(const uint4*)(bp + 16);
        const v16bf b = ub.v;
        acc[j][0] = __builtin_amdgcn_wmma_f32_16x16x32_bf16(
            false, a0, false, b, (short)0, acc[j][0], false, false);
        acc[j][1] = __builtin_amdgcn_wmma_f32_16x16x32_bf16(
            false, a1, false, b, (short)0, acc[j][1], false, false);
      }
    }
#pragma unroll
    for (int j = 0; j < 4; ++j) {
      const int n = ((ntBase + j) << 4) + (lane & 15);
      const float bias = b1e[n];
#pragma unroll
      for (int ms = 0; ms < 2; ++ms) {
#pragma unroll
        for (int v = 0; v < 8; ++v) {
          const int row = (ms << 4) + v + ((lane >> 4) << 3);
          const float t = gelu_exact(acc[j][ms][v] + bias);
          const int addr = row * 8192 + ((((n >> 3) ^ (row & 15)) << 4)) + ((n & 7) << 1);
          *(__bf16*)(hs + addr) = (__bf16)t;
        }
      }
    }
  }
  __syncthreads();

  // ---- Stage 2: out += wgt * (hs @ W2 + b2).  8 waves x 8 n-tiles = 64 tiles.
  for (int ntg = 0; ntg < 2; ++ntg) {
    const int ntBase = wv * 8 + ntg * 4;
    v8f acc[4][2] = {};
    for (int kk = 0; kk < 128; ++kk) {
      const int k0 = kk << 5;
      const v16bf a0 = lds_a_frag(hs, 8192, 0, k0, lane);
      const v16bf a1 = lds_a_frag(hs, 8192, 1, k0, lane);
#pragma unroll
      for (int j = 0; j < 4; ++j) {
        const int n = (ntBase + j) << 4;
        const __bf16* bp = w2e + (size_t)(n + (lane & 15)) * HID_DIM + k0 + ((lane >> 4) << 3);
        union { uint4 q[2]; v16bf v; } ub;
        ub.q[0] = *(const uint4*)bp; ub.q[1] = *(const uint4*)(bp + 16);
        const v16bf b = ub.v;
        acc[j][0] = __builtin_amdgcn_wmma_f32_16x16x32_bf16(
            false, a0, false, b, (short)0, acc[j][0], false, false);
        acc[j][1] = __builtin_amdgcn_wmma_f32_16x16x32_bf16(
            false, a1, false, b, (short)0, acc[j][1], false, false);
      }
    }
#pragma unroll
    for (int j = 0; j < 4; ++j) {
      const int n = ((ntBase + j) << 4) + (lane & 15);
      const float b2v = b2e[n];
#pragma unroll
      for (int ms = 0; ms < 2; ++ms) {
#pragma unroll
        for (int v = 0; v < 8; ++v) {
          const int row = (ms << 4) + v + ((lane >> 4) << 3);
          if (row < rowsValid) {
            const float val = (acc[j][ms][v] + b2v) * tW[row];
            atomicAdd(out + (size_t)tIds[row] * OUT_DIM + n, val);
          }
        }
      }
    }
  }
}

// ---------------------------------------------------------------------------
// Launch
// ---------------------------------------------------------------------------
extern "C" void kernel_launch(void* const* d_in, const int* in_sizes, int n_in,
                              void* d_out, int out_size, void* d_ws, size_t ws_size,
                              hipStream_t stream) {
  const float* x    = (const float*)d_in[0];
  const float* sig  = (const float*)d_in[1];
  const float* tc   = (const float*)d_in[2];
  const float* rs   = (const float*)d_in[3];
  const float* rw1  = (const float*)d_in[4];
  const float* rb1  = (const float*)d_in[5];
  const float* rw2  = (const float*)d_in[6];
  const float* rb2  = (const float*)d_in[7];
  const float* ew1  = (const float*)d_in[8];
  const float* eb1  = (const float*)d_in[9];
  const float* ew2  = (const float*)d_in[10];
  const float* eb2  = (const float*)d_in[11];
  float* out = (float*)d_out;

  // Workspace layout
  char* ws = (char*)d_ws;
  int*    counts = (int*)ws;                                   //    256 B
  int*    tokIds = (int*)(ws + 256);                           //  512 KB
  float*  tokW   = (float*)(ws + 256 + (size_t)NEXP * B_TOK * 4);
  __bf16* w1T    = (__bf16*)(ws + 1048832);                    //   64 MB
  __bf16* w2T    = (__bf16*)(ws + 1048832 + (size_t)NEXP * HID_DIM * IN_DIM * 2);

  hipMemsetAsync(out, 0, (size_t)B_TOK * OUT_DIM * sizeof(float), stream);
  hipMemsetAsync(counts, 0, 256, stream);

  transpose_cvt_kernel<IN_DIM, HID_DIM>
      <<<dim3(HID_DIM / 32, IN_DIM / 32, NEXP), dim3(32, 8), 0, stream>>>(ew1, w1T);
  transpose_cvt_kernel<HID_DIM, OUT_DIM>
      <<<dim3(OUT_DIM / 32, HID_DIM / 32, NEXP), dim3(32, 8), 0, stream>>>(ew2, w2T);

  router_kernel<<<B_TOK / 8, 128, 0, stream>>>(x, sig, tc, rs, rw1, rb1, rw2, rb2,
                                               counts, tokIds, tokW);

  const int smemBytes = 65536 + 262144;   // 320 KB: fits CDNA5's 320KB WGP LDS
  (void)hipFuncSetAttribute((const void*)expert_kernel,
                            hipFuncAttributeMaxDynamicSharedMemorySize, smemBytes);
  expert_kernel<<<dim3(B_TOK / 32, NEXP), 256, smemBytes, stream>>>(
      x, w1T, w2T, eb1, eb2, counts, tokIds, tokW, out);
}
